// LSTM_30897994728319
// MI455X (gfx1250) — compile-verified, matrix operand order checked by reading, not run on hardware
//
#include <hip/hip_runtime.h>
#include <math.h>

// ---------------------------------------------------------------------------
// LSTM single step with zero initial state/hidden:
//   out = tanh( tanh(x@Wxc+bc) * sigmoid(x@Wxi+bi) ) * sigmoid(x@Wxo+bo)
// (forget gate and all Wh terms are dead because h=0, state=0)
//
// Pipeline: f32->bf16 conversion pass (bf16 working set 134MB < 192MB L2),
// then a double-buffered fused 3-GEMM:
//   - A tile staged by the Tensor Data Mover (tensor_load_to_lds, TENSORcnt)
//     with D# LDS padding producing the conflict-free 40-elem row stride
//   - B tiles staged via regs with k-pair-packed b32 transposed stores
//   - 12x v_wmma_f32_16x16x32_bf16 per K-step, fused activation epilogue
// ---------------------------------------------------------------------------

typedef __attribute__((ext_vector_type(16))) __bf16 v16bf_t;
typedef __attribute__((ext_vector_type(8)))  float  v8f_t;
typedef __attribute__((ext_vector_type(4)))  unsigned int v4u_t;
typedef __attribute__((ext_vector_type(8)))  int v8i_t;
typedef __attribute__((ext_vector_type(4)))  int v4i_t;

static constexpr int DIN  = 4096;   // K
static constexpr int DOUT = 4096;   // N (per gate)
static constexpr int NB   = 4096;   // M (batch)

static constexpr int BM = 128;      // macro tile M
static constexpr int BN = 64;       // macro tile N
static constexpr int BK = 32;       // K step == WMMA K for bf16
static constexpr int NITER = DIN / BK;

static constexpr int SA_STRIDE = 40;  // 80B row: 16B-aligned, 20-bank stride (conflict free)
static constexpr int SB_STRIDE = 40;  // same for B ([N][K] layout) -> b128 fragment loads

#if __has_builtin(__builtin_amdgcn_tensor_load_to_lds)
#define HAVE_TDM 1
#else
#define HAVE_TDM 0
#endif

// ------------------------------ conversion ---------------------------------
__global__ __launch_bounds__(256) void cvt_f32_to_bf16(const float* __restrict__ in,
                                                       __bf16* __restrict__ out,
                                                       int n8) {
    int idx = blockIdx.x * blockDim.x + threadIdx.x;
    if (idx >= n8) return;
    const float4* p = (const float4*)in + (size_t)idx * 2;
    float4 a = p[0];
    float4 b = p[1];
    union { uint4 q; __bf16 h[8]; } u;
    u.h[0] = (__bf16)a.x; u.h[1] = (__bf16)a.y;
    u.h[2] = (__bf16)a.z; u.h[3] = (__bf16)a.w;
    u.h[4] = (__bf16)b.x; u.h[5] = (__bf16)b.y;
    u.h[6] = (__bf16)b.z; u.h[7] = (__bf16)b.w;
    ((uint4*)out)[idx] = u.q;
}

// -------------------- TDM: A tile (128x32 bf16) -> LDS ---------------------
#if HAVE_TDM
__device__ __forceinline__ void tdm_load_tile_a(const __bf16* gsrc, unsigned lds_addr) {
    unsigned long long ga = (unsigned long long)(uintptr_t)gsrc;
    // D# group0: count=1 | lds_addr | global_addr[56:0] | type=2
    v4u_t g0 = { 1u,
                 lds_addr,
                 (unsigned)(ga & 0xFFFFFFFFu),
                 (unsigned)(((ga >> 32) & 0x1FFFFFFu) | (2u << 30)) };
    // D# group1: data_size=2B, pad_enable, pad_interval=16dw, pad_amount=4dw
    //            tensor_dim0=4096, tensor_dim1=4096, tile 32x128,
    //            tensor_dim0_stride=4096, tensor_dim1_stride=4096
    v8i_t g1 = { (int)((1u << 16) | (1u << 20) | (3u << 22) | (3u << 25)),
                 (int)(0x1000u << 16),    // tensor_dim0 lo16 @ [63:48]
                 (int)(0x1000u << 16),    // dim0 hi16=0 | tensor_dim1 lo16 @ [95:80]
                 (int)(32u << 16),        // dim1 hi16=0 | tile_dim0=32 @ [127:112]
                 (int)128,                // tile_dim1=128, tile_dim2=0
                 (int)4096,               // tensor_dim0_stride[31:0]
                 (int)(0x1000u << 16),    // stride0 hi=0 | tensor_dim1_stride lo16
                 0 };
    v4i_t g2 = { 0, 0, 0, 0 };
    v4i_t g3 = { 0, 0, 0, 0 };
#if defined(__clang_major__) && (__clang_major__ >= 23)
    v8i_t g4 = { 0, 0, 0, 0, 0, 0, 0, 0 };
    __builtin_amdgcn_tensor_load_to_lds(g0, g1, g2, g3, g4, 0);
#else
    __builtin_amdgcn_tensor_load_to_lds(g0, g1, g2, g3, 0);
#endif
}
__device__ __forceinline__ void tdm_wait0() {
#if __has_builtin(__builtin_amdgcn_s_wait_tensorcnt)
    __builtin_amdgcn_s_wait_tensorcnt((short)0);
#else
    asm volatile("s_wait_tensorcnt 0x0" ::: "memory");
#endif
}
#endif

__device__ __forceinline__ float fast_tanh(float x) {
    float e = __expf(2.0f * x);
    return 1.0f - 2.0f / (e + 1.0f);
}

// ------------------------------ fused 3-GEMM -------------------------------
__global__ __launch_bounds__(256) void lstm_gemm_bf16(
    const __bf16* __restrict__ xb,
    const __bf16* __restrict__ wib,   // Wxi bf16 [K][N]
    const __bf16* __restrict__ wcb,   // Wxc bf16 [K][N]
    const __bf16* __restrict__ wob,   // Wxo bf16 [K][N]
    const float*  __restrict__ bi,
    const float*  __restrict__ bc,
    const float*  __restrict__ bo,
    float* __restrict__ out)
{
    __shared__ __attribute__((aligned(16))) __bf16 sA[2][BM * SA_STRIDE];
    __shared__ __attribute__((aligned(16))) __bf16 sB[2][3][BN * SB_STRIDE];

    const int t    = threadIdx.x;
    const int lane = t & 31;
    const int wave = t >> 5;        // 0..7
    const int wm   = wave & 3;      // M sub-block (each 32 rows)
    const int wn   = wave >> 2;     // N sub-block (each 32 cols)
    const int m0   = blockIdx.y * BM;
    const int n0   = blockIdx.x * BN;

    const __bf16* wptr[3] = { wib, wcb, wob };

    // accumulators: 3 gates x 2x2 WMMA tiles
    v8f_t acc[3][2][2];
    v8f_t zero = {};
#pragma unroll
    for (int g = 0; g < 3; ++g)
#pragma unroll
        for (int i2 = 0; i2 < 2; ++i2)
#pragma unroll
            for (int j2 = 0; j2 < 2; ++j2) acc[g][i2][j2] = zero;

    // per-lane K indices of the 8 packed bf16 pairs (ISA 16-bit A/B layout);
    // with 40-elem LDS row stride these merge into 2x ds_load_b128 per frag.
    const int khalf = (lane >> 4) * 8;
    int kidx[8];
#pragma unroll
    for (int j = 0; j < 8; ++j)
        kidx[j] = ((j < 4) ? (2 * j) : (16 + 2 * (j - 4))) + khalf;

    const int am = lane & 15;

    // B staging coords: thread handles rows {2kp, 2kp+1} x 4 cols per gate
    const int kp = t >> 4;          // 0..15 -> k pair base 2*kp
    const int c4 = t & 15;          // 0..15 -> cols 4*c4 .. 4*c4+3

#if HAVE_TDM
    const unsigned sA_lds0 = (unsigned)(uintptr_t)(void*)&sA[0][0];
    const unsigned sA_lds1 = (unsigned)(uintptr_t)(void*)&sA[1][0];
#endif

    union U2 { uint2 v; unsigned short h[4]; };

    // -------- prologue: stage tile 0 into buffer 0 --------
#if HAVE_TDM
    if (wave == 0)
        tdm_load_tile_a(xb + (size_t)m0 * DIN, sA_lds0);
#else
#pragma unroll
    for (int s = 0; s < 2; ++s) {
        int u = t + s * 256;
        int row = u >> 2, c8 = u & 3;
        uint4 v = *((const uint4*)(xb + (size_t)(m0 + row) * DIN) + c8);
        ((uint4*)sA[0])[row * 5 + c8] = v;
    }
#endif
#pragma unroll
    for (int g = 0; g < 3; ++g) {
        U2 r0, r1;
        r0.v = *(const uint2*)(wptr[g] + (size_t)(2 * kp) * DOUT + n0 + 4 * c4);
        r1.v = *(const uint2*)(wptr[g] + (size_t)(2 * kp + 1) * DOUT + n0 + 4 * c4);
#pragma unroll
        for (int e = 0; e < 4; ++e) {
            unsigned pk = (unsigned)r0.h[e] | ((unsigned)r1.h[e] << 16);
            *(unsigned*)&sB[0][g][(4 * c4 + e) * SB_STRIDE + 2 * kp] = pk;
        }
    }
#if HAVE_TDM
    if (wave == 0) tdm_wait0();
#endif
    __syncthreads();

    // -------- main double-buffered loop --------
    for (int it = 0; it < NITER; ++it) {
        const int p = it & 1;
        const int q = p ^ 1;
        const bool more = (it + 1 < NITER);
        const int kkn = (it + 1) * BK;

        // issue next A tile DMA + next B global loads (overlap with compute)
        U2 br0[3], br1[3];
        if (more) {
#if HAVE_TDM
            if (wave == 0)
                tdm_load_tile_a(xb + (size_t)m0 * DIN + kkn, q ? sA_lds1 : sA_lds0);
#endif
#pragma unroll
            for (int g = 0; g < 3; ++g) {
                br0[g].v = *(const uint2*)(wptr[g] + (size_t)(kkn + 2 * kp) * DOUT + n0 + 4 * c4);
                br1[g].v = *(const uint2*)(wptr[g] + (size_t)(kkn + 2 * kp + 1) * DOUT + n0 + 4 * c4);
            }
        }

        // ---- compute tile `it` from buffer p ----
        union Frag { v16bf_t v; unsigned int u[8]; };
        Frag afrag[2];
#pragma unroll
        for (int i2 = 0; i2 < 2; ++i2) {
            int mrow = wm * 32 + i2 * 16 + am;
#pragma unroll
            for (int j = 0; j < 8; ++j)
                afrag[i2].u[j] = *(const unsigned int*)(sA[p] + mrow * SA_STRIDE + kidx[j]);
        }
#pragma unroll
        for (int g = 0; g < 3; ++g) {
#pragma unroll
            for (int j2 = 0; j2 < 2; ++j2) {
                Frag bfrag;
                int ncol = wn * 32 + j2 * 16 + am;
#pragma unroll
                for (int j = 0; j < 8; ++j)
                    bfrag.u[j] = *(const unsigned int*)(sB[p][g] + ncol * SB_STRIDE + kidx[j]);
#pragma unroll
                for (int i2 = 0; i2 < 2; ++i2)
                    acc[g][i2][j2] = __builtin_amdgcn_wmma_f32_16x16x32_bf16(
                        false, afrag[i2].v, false, bfrag.v,
                        (short)0, acc[g][i2][j2], false, false);
            }
        }

        // ---- drain next-tile staging into buffer q ----
        if (more) {
#if !HAVE_TDM
#pragma unroll
            for (int s = 0; s < 2; ++s) {
                int u = t + s * 256;
                int row = u >> 2, c8 = u & 3;
                uint4 v = *((const uint4*)(xb + (size_t)(m0 + row) * DIN + kkn) + c8);
                ((uint4*)sA[q])[row * 5 + c8] = v;
            }
#endif
#pragma unroll
            for (int g = 0; g < 3; ++g) {
#pragma unroll
                for (int e = 0; e < 4; ++e) {
                    unsigned pk = (unsigned)br0[g].h[e] | ((unsigned)br1[g].h[e] << 16);
                    *(unsigned*)&sB[q][g][(4 * c4 + e) * SB_STRIDE + 2 * kp] = pk;
                }
            }
#if HAVE_TDM
            if (wave == 0) tdm_wait0();
#endif
        }
        __syncthreads();
    }

    // ---- fused epilogue: bias + sigmoid/tanh + store ----
    // C/D layout: VGPR r -> row r + 8*(lane/16); lane&15 -> column
    const int nlocal = lane & 15;
    const int mhalf  = (lane >> 4) * 8;
#pragma unroll
    for (int j2 = 0; j2 < 2; ++j2) {
        int col = n0 + wn * 32 + j2 * 16 + nlocal;
        float bbi = bi[col], bbc = bc[col], bbo = bo[col];
#pragma unroll
        for (int i2 = 0; i2 < 2; ++i2) {
            int rowbase = m0 + wm * 32 + i2 * 16 + mhalf;
#pragma unroll
            for (int r = 0; r < 8; ++r) {
                float gi = acc[0][i2][j2][r] + bbi;
                float gc = acc[1][i2][j2][r] + bbc;
                float go = acc[2][i2][j2][r] + bbo;
                float si = 1.0f / (1.0f + __expf(-gi));
                float so = 1.0f / (1.0f + __expf(-go));
                float tc = fast_tanh(gc);
                out[(size_t)(rowbase + r) * DOUT + col] = fast_tanh(tc * si) * so;
            }
        }
    }
}

// ------------------------------ launch -------------------------------------
extern "C" void kernel_launch(void* const* d_in, const int* in_sizes, int n_in,
                              void* d_out, int out_size, void* d_ws, size_t ws_size,
                              hipStream_t stream) {
    // setup_inputs order:
    // 0:x 1:Wxf 2:bxf 3:Whf 4:Wxi 5:bxi 6:Whi 7:Wxc 8:bxc 9:Whc 10:Wxo 11:bxo 12:Who
    const float* x   = (const float*)d_in[0];
    const float* Wxi = (const float*)d_in[4];
    const float* bxi = (const float*)d_in[5];
    const float* Wxc = (const float*)d_in[7];
    const float* bxc = (const float*)d_in[8];
    const float* Wxo = (const float*)d_in[10];
    const float* bxo = (const float*)d_in[11];
    float* out = (float*)d_out;

    const size_t N = (size_t)DIN * DOUT;      // 16.7M elems per matrix (== NB*DIN)
    __bf16* xb  = (__bf16*)d_ws;
    __bf16* wib = xb  + N;
    __bf16* wcb = wib + N;
    __bf16* wob = wcb + N;

    const int n8 = (int)(N / 8);
    dim3 cgrid((n8 + 255) / 256), cblk(256);
    cvt_f32_to_bf16<<<cgrid, cblk, 0, stream>>>(x,   xb,  n8);
    cvt_f32_to_bf16<<<cgrid, cblk, 0, stream>>>(Wxi, wib, n8);
    cvt_f32_to_bf16<<<cgrid, cblk, 0, stream>>>(Wxc, wcb, n8);
    cvt_f32_to_bf16<<<cgrid, cblk, 0, stream>>>(Wxo, wob, n8);

    dim3 grid(DOUT / BN, NB / BM);            // (64, 32)
    lstm_gemm_bf16<<<grid, 256, 0, stream>>>(xb, wib, wcb, wob, bxi, bxc, bxo, out);
}
